// GraphMultiHeadAttentionLayer_8169027797154
// MI455X (gfx1250) — matrix operand order repaired and checked by reading, not compiled, not run on hardware
//
#include <hip/hip_runtime.h>
#include <hip/hip_bf16.h>

// ---------------------------------------------------------------------------
// GraphTransformerConv (HEADS=1, DIN=DOUT=128, DEDGE=64) + LN + gated residual
// N=50000 nodes, E=800000 edges.
// GEMMs via v_wmma_f32_16x16x32_f16 with pre-packed f16 operands:
//   - weights packed once into WMMA B layout (180KB, L2-resident)
//   - x / edge_attr packed once into WMMA A layout (f16)
// so the GEMM inner loop is pure b128 loads + wmma.
// ---------------------------------------------------------------------------

#define GN 50000
#define GE 800000
#define GDIN 128
#define GDOUT 128
#define GDEDGE 64

typedef _Float16 v16h __attribute__((ext_vector_type(16)));
typedef float    v8f  __attribute__((ext_vector_type(8)));
typedef _Float16 half4 __attribute__((ext_vector_type(4)));

// packed-weight offsets in v16h (32B) units: 128x128 weight = 1024 units
#define WOFF_Q    0
#define WOFF_K    1024
#define WOFF_V    2048
#define WOFF_SKIP 3072
#define WOFF_RES  4096
#define WOFF_E    5120          // We: 64x128 = 512 units
#define WPACK_UNITS (5120 + 512)

// monotonic float<->uint encoding for atomicMax on signed floats
__device__ __forceinline__ unsigned enc_f32(float f) {
    unsigned u = __float_as_uint(f);
    return (u & 0x80000000u) ? ~u : (u | 0x80000000u);
}
__device__ __forceinline__ float dec_f32(unsigned u) {
    return __uint_as_float((u & 0x80000000u) ? (u & 0x7FFFFFFFu) : ~u);
}

// ---------------------------------------------------------------------------
// P1: pack one f32 weight [Kdim x 128] into f16 WMMA B layout.
// Per wave = one (nt, kt) 32x16 B tile; lane stores one contiguous v16h.
// B layout: lane L holds column L&15; element i -> k = i + 16*(L>>4).
// dst unit index: (nt*KT + kt)*32 + lane
// ---------------------------------------------------------------------------
__global__ void pack_weightB_kernel(const float* __restrict__ W, int Kdim,
                                    _Float16* __restrict__ dst) {
    const int lane = threadIdx.x & 31;
    const int KT = Kdim >> 5;
    const int tile = blockIdx.x * (blockDim.x >> 5) + (threadIdx.x >> 5);
    if (tile >= 8 * KT) return;
    const int nt = tile / KT, kt = tile % KT;
    const int hi = lane >> 4, lo = lane & 15;
    v16h t;
#pragma unroll
    for (int i = 0; i < 16; ++i) {
        int kk = kt * 32 + i + 16 * hi;
        t[i] = (_Float16)W[(size_t)kk * GDOUT + nt * 16 + lo];
    }
    ((v16h*)dst)[(size_t)tile * 32 + lane] = t;
}

// ---------------------------------------------------------------------------
// P2: pack f32 activation [M x Kdim] into f16 WMMA A layout.
// Per wave = one (mtile, kt) 16x32 A tile; lane stores one contiguous v16h.
// A layout: lane L holds row L&15; element i -> k = i + 16*(L>>4).
// dst unit index: tile*32 + lane   (tile = mtile*KT + kt)
// ---------------------------------------------------------------------------
__global__ void pack_A_kernel(const float* __restrict__ src, int Kdim, int ntiles,
                              _Float16* __restrict__ dst) {
    const int lane = threadIdx.x & 31;
    const int tile = blockIdx.x * (blockDim.x >> 5) + (threadIdx.x >> 5);
    if (tile >= ntiles) return;
    const int KT = Kdim >> 5;
    const int mtile = tile / KT, kt = tile % KT;
    const int hi = lane >> 4, lo = lane & 15;
    const float* row = src + (size_t)(mtile * 16 + lo) * Kdim;
    v16h t;
#pragma unroll
    for (int i = 0; i < 16; ++i) {
        int kk = kt * 32 + i + 16 * hi;
        t[i] = (_Float16)row[kk];
    }
    ((v16h*)dst)[(size_t)tile * 32 + lane] = t;
}

// ---------------------------------------------------------------------------
// K1: node GEMMs. One wave = 16 rows x one whole 128-col weight (8 n-tiles),
// A tile loaded once (b128s) and reused 8x; inner loop = b128 loads + wmma.
// tiles = 3125 m-tiles x 5 weights.
// ---------------------------------------------------------------------------
__global__ void node_gemm_kernel(const _Float16* __restrict__ xp,
                                 const _Float16* __restrict__ wp,
                                 const float* __restrict__ bq,
                                 const float* __restrict__ bk,
                                 const float* __restrict__ bv,
                                 const float* __restrict__ bskip,
                                 const float* __restrict__ bres,
                                 float* __restrict__ q, float* __restrict__ k,
                                 float* __restrict__ v, float* __restrict__ skip,
                                 float* __restrict__ res) {
    const int lane = threadIdx.x & 31;
    const int tile = blockIdx.x * (blockDim.x >> 5) + (threadIdx.x >> 5);
    const int NTM = GN / 16;                       // 3125
    if (tile >= NTM * 5) return;
    const int mtile = tile / 5;
    const int wsel  = tile % 5;
    const int hi = lane >> 4, lo = lane & 15;

    const v16h* xv = (const v16h*)xp;
    const v16h* wv = (const v16h*)wp;

    v16h a[4];
#pragma unroll
    for (int kt = 0; kt < 4; ++kt)
        a[kt] = xv[(size_t)(mtile * 4 + kt) * 32 + lane];

    const float* bias_p; float* out; size_t wbase;
    switch (wsel) {
        case 0:  bias_p = bq;    out = q;    wbase = WOFF_Q;    break;
        case 1:  bias_p = bk;    out = k;    wbase = WOFF_K;    break;
        case 2:  bias_p = bv;    out = v;    wbase = WOFF_V;    break;
        case 3:  bias_p = bskip; out = skip; wbase = WOFF_SKIP; break;
        default: bias_p = bres;  out = res;  wbase = WOFF_RES;  break;
    }

#pragma unroll
    for (int nt = 0; nt < 8; ++nt) {
        v8f acc = {0.f, 0.f, 0.f, 0.f, 0.f, 0.f, 0.f, 0.f};
#pragma unroll
        for (int kt = 0; kt < 4; ++kt) {
            v16h b = wv[wbase + (size_t)(nt * 4 + kt) * 32 + lane];
            acc = __builtin_amdgcn_wmma_f32_16x16x32_f16(
                false, a[kt], false, b, (short)0, acc, false, false);
        }
        const int col = nt * 16 + lo;
        const float bias = bias_p[col];
#pragma unroll
        for (int r = 0; r < 8; ++r) {
            int m = mtile * 16 + r + 8 * hi;       // ISA C/D layout
            out[(size_t)m * GDOUT + col] = acc[r] + bias;
        }
    }
}

// ---------------------------------------------------------------------------
// K2: gate[m] = sigmoid(x[m] . Wgate + bgate)
// ---------------------------------------------------------------------------
__global__ void gate_kernel(const float* __restrict__ x,
                            const float* __restrict__ Wgate,
                            const float* __restrict__ bgate,
                            float* __restrict__ gate) {
    int m = blockIdx.x * blockDim.x + threadIdx.x;
    if (m >= GN) return;
    const float4* xr = (const float4*)(x + (size_t)m * GDIN);
    const float4* wr = (const float4*)Wgate;
    float z = bgate[0];
#pragma unroll 8
    for (int i = 0; i < GDIN / 4; ++i) {
        float4 xv = xr[i], wv = wr[i];
        z += xv.x * wv.x + xv.y * wv.y + xv.z * wv.z + xv.w * wv.w;
    }
    gate[m] = 1.0f / (1.0f + __expf(-z));
}

// ---------------------------------------------------------------------------
// K3: edge GEMM  e = edge_attr @ We (E x 64 x 128), output f16.
// One wave = 16 edges x all 128 cols; A loaded once (2 k-tiles), 16 WMMAs.
// ---------------------------------------------------------------------------
__global__ void edge_gemm_kernel(const _Float16* __restrict__ ep,
                                 const _Float16* __restrict__ wp,
                                 _Float16* __restrict__ e_half) {
    const int lane = threadIdx.x & 31;
    const int etile = blockIdx.x * (blockDim.x >> 5) + (threadIdx.x >> 5);
    if (etile >= GE / 16) return;
    const int hi = lane >> 4, lo = lane & 15;

    const v16h* av = (const v16h*)ep;
    const v16h* wv = (const v16h*)wp;

    v16h a[2];
#pragma unroll
    for (int kt = 0; kt < 2; ++kt)
        a[kt] = av[(size_t)(etile * 2 + kt) * 32 + lane];

#pragma unroll
    for (int nt = 0; nt < 8; ++nt) {
        v8f acc = {0.f, 0.f, 0.f, 0.f, 0.f, 0.f, 0.f, 0.f};
#pragma unroll
        for (int kt = 0; kt < 2; ++kt) {
            v16h b = wv[WOFF_E + (size_t)(nt * 2 + kt) * 32 + lane];
            acc = __builtin_amdgcn_wmma_f32_16x16x32_f16(
                false, a[kt], false, b, (short)0, acc, false, false);
        }
        const int col = nt * 16 + lo;
#pragma unroll
        for (int r = 0; r < 8; ++r) {
            int m = etile * 16 + r + 8 * hi;
            e_half[(size_t)m * GDOUT + col] = (_Float16)acc[r];
        }
    }
}

// ---------------------------------------------------------------------------
// K4: per-edge alpha = q[dst].(k[src]+e)/sqrt(C); atomicMax segment max.
// q/k gathers are L2-resident (25.6MB each vs 192MB L2).
// ---------------------------------------------------------------------------
__global__ void alpha_kernel(const int* __restrict__ src, const int* __restrict__ dst,
                             const float* __restrict__ q, const float* __restrict__ k,
                             const _Float16* __restrict__ e_half,
                             float* __restrict__ alpha, unsigned* __restrict__ amax) {
    const int lane = threadIdx.x & 31;
    const int e = blockIdx.x * (blockDim.x >> 5) + (threadIdx.x >> 5);
    if (e >= GE) return;
    const int s = src[e], d = dst[e];
    float4 qv = ((const float4*)(q + (size_t)d * GDOUT))[lane];
    float4 kv = ((const float4*)(k + (size_t)s * GDOUT))[lane];
    half4  ev = ((const half4*)(e_half + (size_t)e * GDOUT))[lane];
    float p = qv.x * (kv.x + (float)ev[0]) + qv.y * (kv.y + (float)ev[1]) +
              qv.z * (kv.z + (float)ev[2]) + qv.w * (kv.w + (float)ev[3]);
#pragma unroll
    for (int m = 16; m >= 1; m >>= 1) p += __shfl_xor(p, m, 32);
    if (lane == 0) {
        float al = p * 0.08838834764831845f;   // 1/sqrt(128)
        alpha[e] = al;
        atomicMax(amax + d, enc_f32(al));
    }
}

// ---------------------------------------------------------------------------
// K5: exp(alpha - amax[dst]); accumulate segment denominator.
// ---------------------------------------------------------------------------
__global__ void expsum_kernel(const int* __restrict__ dst,
                              float* __restrict__ alpha,
                              const unsigned* __restrict__ amax,
                              float* __restrict__ denom) {
    int e = blockIdx.x * blockDim.x + threadIdx.x;
    if (e >= GE) return;
    int d = dst[e];
    float ex = __expf(alpha[e] - dec_f32(amax[d]));
    alpha[e] = ex;
    atomicAdd(denom + d, ex);
}

// ---------------------------------------------------------------------------
// K6: scatter  attnout[dst] += p * (v[src] + e).  One wave per edge.
// ---------------------------------------------------------------------------
__global__ void scatter_kernel(const int* __restrict__ src, const int* __restrict__ dst,
                               const float* __restrict__ v,
                               const _Float16* __restrict__ e_half,
                               const float* __restrict__ alpha,
                               const float* __restrict__ denom,
                               float* __restrict__ attnout) {
    const int lane = threadIdx.x & 31;
    const int e = blockIdx.x * (blockDim.x >> 5) + (threadIdx.x >> 5);
    if (e >= GE) return;
    const int s = src[e], d = dst[e];
    const float p = alpha[e] / denom[d];
    float4 vv = ((const float4*)(v + (size_t)s * GDOUT))[lane];
    half4  ev = ((const half4*)(e_half + (size_t)e * GDOUT))[lane];
    float* o = attnout + (size_t)d * GDOUT + lane * 4;
    atomicAdd(o + 0, p * (vv.x + (float)ev[0]));
    atomicAdd(o + 1, p * (vv.y + (float)ev[1]));
    atomicAdd(o + 2, p * (vv.z + (float)ev[2]));
    atomicAdd(o + 3, p * (vv.w + (float)ev[3]));
}

// ---------------------------------------------------------------------------
// K7: out = gate * relu(LN(attn + skip)) + (1-gate) * res.  One wave per node.
// ---------------------------------------------------------------------------
__global__ void final_kernel(const float* __restrict__ attnout,
                             const float* __restrict__ skip,
                             const float* __restrict__ res,
                             const float* __restrict__ gate,
                             const float* __restrict__ ln_g,
                             const float* __restrict__ ln_b,
                             float* __restrict__ out) {
    const int lane = threadIdx.x & 31;
    const int m = blockIdx.x * (blockDim.x >> 5) + (threadIdx.x >> 5);
    if (m >= GN) return;
    float4 av = ((const float4*)(attnout + (size_t)m * GDOUT))[lane];
    float4 sv = ((const float4*)(skip    + (size_t)m * GDOUT))[lane];
    float s0 = av.x + sv.x, s1 = av.y + sv.y, s2 = av.z + sv.z, s3 = av.w + sv.w;
    float sum = s0 + s1 + s2 + s3;
#pragma unroll
    for (int msk = 16; msk >= 1; msk >>= 1) sum += __shfl_xor(sum, msk, 32);
    const float mu = sum * (1.0f / GDOUT);
    float d0 = s0 - mu, d1 = s1 - mu, d2 = s2 - mu, d3 = s3 - mu;
    float sq = d0 * d0 + d1 * d1 + d2 * d2 + d3 * d3;
#pragma unroll
    for (int msk = 16; msk >= 1; msk >>= 1) sq += __shfl_xor(sq, msk, 32);
    const float inv = rsqrtf(sq * (1.0f / GDOUT) + 1e-5f);
    float4 gv = ((const float4*)ln_g)[lane];
    float4 bv = ((const float4*)ln_b)[lane];
    float y0 = fmaxf(d0 * inv * gv.x + bv.x, 0.f);
    float y1 = fmaxf(d1 * inv * gv.y + bv.y, 0.f);
    float y2 = fmaxf(d2 * inv * gv.z + bv.z, 0.f);
    float y3 = fmaxf(d3 * inv * gv.w + bv.w, 0.f);
    const float g = gate[m];
    float4 rv = ((const float4*)(res + (size_t)m * GDOUT))[lane];
    float4 o;
    o.x = g * y0 + (1.f - g) * rv.x;
    o.y = g * y1 + (1.f - g) * rv.y;
    o.z = g * y2 + (1.f - g) * rv.z;
    o.w = g * y3 + (1.f - g) * rv.w;
    ((float4*)(out + (size_t)m * GDOUT))[lane] = o;
}

// ---------------------------------------------------------------------------
extern "C" void kernel_launch(void* const* d_in, const int* in_sizes, int n_in,
                              void* d_out, int out_size, void* d_ws, size_t ws_size,
                              hipStream_t stream) {
    (void)in_sizes; (void)n_in; (void)out_size; (void)ws_size;
    const float* x         = (const float*)d_in[0];
    const int*   ei        = (const int*)  d_in[1];
    const float* edge_attr = (const float*)d_in[2];
    const float* Wq    = (const float*)d_in[3];  const float* bq    = (const float*)d_in[4];
    const float* Wk    = (const float*)d_in[5];  const float* bk    = (const float*)d_in[6];
    const float* Wv    = (const float*)d_in[7];  const float* bv    = (const float*)d_in[8];
    const float* We    = (const float*)d_in[9];
    const float* Wskip = (const float*)d_in[10]; const float* bskip = (const float*)d_in[11];
    const float* ln_g  = (const float*)d_in[12]; const float* ln_b  = (const float*)d_in[13];
    const float* Wres  = (const float*)d_in[14]; const float* bres  = (const float*)d_in[15];
    const float* Wgate = (const float*)d_in[16]; const float* bgate = (const float*)d_in[17];
    const int* src = ei;           // edge_index[0]
    const int* dst = ei + GE;      // edge_index[1]
    float* out = (float*)d_out;

    // workspace carve-up (256B aligned)
    char* w = (char*)d_ws;
    size_t off = 0;
    auto carve = [&](size_t bytes) -> void* {
        void* p = (void*)(w + off);
        off = (off + bytes + 255) & ~(size_t)255;
        return p;
    };
    const size_t rowbytes = (size_t)GN * GDOUT * sizeof(float);
    float*     q       = (float*)    carve(rowbytes);
    float*     k       = (float*)    carve(rowbytes);
    float*     v       = (float*)    carve(rowbytes);
    float*     skip    = (float*)    carve(rowbytes);
    float*     res     = (float*)    carve(rowbytes);
    float*     attnout = (float*)    carve(rowbytes);
    float*     gate    = (float*)    carve((size_t)GN * sizeof(float));
    float*     alpha   = (float*)    carve((size_t)GE * sizeof(float));
    unsigned*  amax    = (unsigned*) carve((size_t)GN * sizeof(unsigned));
    float*     denom   = (float*)    carve((size_t)GN * sizeof(float));
    _Float16*  e_half  = (_Float16*) carve((size_t)GE * GDOUT * sizeof(_Float16));
    _Float16*  wpack   = (_Float16*) carve((size_t)WPACK_UNITS * 16 * sizeof(_Float16));
    _Float16*  xp      = (_Float16*) carve((size_t)GN * GDIN * sizeof(_Float16));
    _Float16*  epk     = (_Float16*) carve((size_t)GE * GDEDGE * sizeof(_Float16));

    // zero segment-reduction state
    hipMemsetAsync(attnout, 0, rowbytes, stream);
    hipMemsetAsync(amax, 0, (size_t)GN * sizeof(unsigned), stream);   // enc(-huge)
    hipMemsetAsync(denom, 0, (size_t)GN * sizeof(float), stream);

    // P1: pack weights into WMMA B layout (tiny, one-shot)
    pack_weightB_kernel<<<4, 256, 0, stream>>>(Wq,    128, wpack + (size_t)WOFF_Q    * 16);
    pack_weightB_kernel<<<4, 256, 0, stream>>>(Wk,    128, wpack + (size_t)WOFF_K    * 16);
    pack_weightB_kernel<<<4, 256, 0, stream>>>(Wv,    128, wpack + (size_t)WOFF_V    * 16);
    pack_weightB_kernel<<<4, 256, 0, stream>>>(Wskip, 128, wpack + (size_t)WOFF_SKIP * 16);
    pack_weightB_kernel<<<4, 256, 0, stream>>>(Wres,  128, wpack + (size_t)WOFF_RES  * 16);
    pack_weightB_kernel<<<2, 256, 0, stream>>>(We,     64, wpack + (size_t)WOFF_E    * 16);
    // P2: pack activations into WMMA A layout
    {
        int xtiles = (GN / 16) * 4;                 // 12500
        pack_A_kernel<<<(xtiles + 7) / 8, 256, 0, stream>>>(x, GDIN, xtiles, xp);
        int etiles = (GE / 16) * 2;                 // 100000
        pack_A_kernel<<<(etiles + 7) / 8, 256, 0, stream>>>(edge_attr, GDEDGE, etiles, epk);
    }

    // K1: node GEMMs (3125 m-tiles x 5 weights, 8 waves/block)
    {
        int tiles = (GN / 16) * 5;                  // 15625
        node_gemm_kernel<<<(tiles + 7) / 8, 256, 0, stream>>>(
            xp, wpack, bq, bk, bv, bskip, bres, q, k, v, skip, res);
    }
    // K2: gate
    gate_kernel<<<(GN + 255) / 256, 256, 0, stream>>>(x, Wgate, bgate, gate);
    // K3: edge GEMM -> e (f16)
    edge_gemm_kernel<<<(GE / 16 + 7) / 8, 256, 0, stream>>>(epk, wpack, e_half);
    // K4: alpha + segment max
    alpha_kernel<<<(GE + 7) / 8, 256, 0, stream>>>(src, dst, q, k, e_half, alpha, amax);
    // K5: exp + segment sum
    expsum_kernel<<<(GE + 255) / 256, 256, 0, stream>>>(dst, alpha, amax, denom);
    // K6: weighted scatter
    scatter_kernel<<<(GE + 7) / 8, 256, 0, stream>>>(src, dst, v, e_half, alpha, denom, attnout);
    // K7: LN + relu + gated residual
    final_kernel<<<(GN + 7) / 8, 256, 0, stream>>>(attnout, skip, res, gate, ln_g, ln_b, out);
}
